// FourierConv2D_62234076119189
// MI455X (gfx1250) — matrix operand order repaired and checked by reading, not compiled, not run on hardware
//
#include <hip/hip_runtime.h>
#include <hip/hip_bf16.h>
#include <math.h>

// ---------------------------------------------------------------------------
// FourierConv2D on gfx1250 via DFT-as-GEMM with v_wmma_f32_16x16x32_f16.
// N = 511 (padded to 512 in all workspace buffers; pad row/col of the twiddle
// matrices are zero so padded garbage annihilates).
// Workspace need: ~79.2 MB (fp16 planes).
// ---------------------------------------------------------------------------

#define NDFT 511
#define NP   512          // padded
#define HH   256
#define BB   4
#define CIN  4
#define COUT 8

#define PLANE_D (NP * HH)      // 512 x 256 = 131072 (D fwd twiddle, A-layout)
#define PLANE_E (HH * NP)      // 256 x 512 = 131072 (E inv twiddle, A & Bt layout)
#define PLANE_X (HH * HH)      // 256 x 256 =  65536 (transposed input, Bt layout)
#define PLANE_T (NP * HH)      // 512 x 256 = 131072 (T, row-major [u][x])
#define PLANE_F (NP * NP)      // 512 x 512 = 262144 (F, row-major [u][v])
#define PLANE_G (NP * NP)      // 512 x 512 = 262144 (G, row-major [u][v])
#define PLANE_Q (HH * NP)      // 256 x 512 = 131072 (Qt, row-major [x][u])

typedef __attribute__((ext_vector_type(16))) _Float16 v16h;
typedef __attribute__((ext_vector_type(8)))  _Float16 v8h;
typedef __attribute__((ext_vector_type(8)))  float    v8f;

// ---------------------------------------------------------------------------
// Twiddle generation:
//   Dfwd[u][y] = e^{-2pi i u y / 511}          (512x256, row 511 = 0)
//   E[r][c]    = e^{+2pi i c (r+127)/511}/32   (256x512, col 511 = 0)
// Also emits pre-negated imaginary planes Dni = -Di, Eni = -Ei so the GEMM
// kernel never needs a runtime negate (the "minus" product of the complex
// multiply just points at the negated plane).
// ---------------------------------------------------------------------------
__global__ void twiddle_kernel(_Float16* __restrict__ Dr, _Float16* __restrict__ Di,
                               _Float16* __restrict__ Dni,
                               _Float16* __restrict__ Er, _Float16* __restrict__ Ei,
                               _Float16* __restrict__ Eni) {
    int t = blockIdx.x * 256 + threadIdx.x;   // 0 .. 131071
    {   // Dfwd: row = t>>8 (0..511), col = t & 255
        int row = t >> 8, col = t & 255;
        float cr = 0.0f, ci = 0.0f;
        if (row < NDFT) {
            int m = (row * col) % NDFT;
            float ang = -6.283185307179586f * (float)m / (float)NDFT;
            float s, c; sincosf(ang, &s, &c);
            cr = c; ci = s;
        }
        Dr[t] = (_Float16)cr; Di[t] = (_Float16)ci; Dni[t] = (_Float16)(-ci);
    }
    {   // E: row = t>>9 (0..255), col = t & 511
        int row = t >> 9, col = t & 511;
        float cr = 0.0f, ci = 0.0f;
        if (col < NDFT) {
            int m = (col * (row + 127)) % NDFT;
            float ang = 6.283185307179586f * (float)m / (float)NDFT;
            float s, c; sincosf(ang, &s, &c);
            cr = c * 0.03125f; ci = s * 0.03125f;   // fold 1/32 (power of two)
        }
        Er[t] = (_Float16)cr; Ei[t] = (_Float16)ci; Eni[t] = (_Float16)(-ci);
    }
}

// ---------------------------------------------------------------------------
// Input prep: f32 image (256x256, row-major [y][x]) -> fp16 transposed [x][y]
// (Bt layout for stage 1). Standard LDS tile transpose.
// ---------------------------------------------------------------------------
__global__ void prep_transpose_kernel(const float* __restrict__ im,
                                      _Float16* __restrict__ Xt) {
    __shared__ float tile[32][33];
    const int img = blockIdx.z;                // b*CIN + c
    const int x0 = blockIdx.x * 32, y0 = blockIdx.y * 32;
    const int tx = threadIdx.x, ty = threadIdx.y;   // 32 x 8
    const float* src = im + (size_t)img * PLANE_X;
    #pragma unroll
    for (int j = 0; j < 32; j += 8)
        tile[ty + j][tx] = src[(size_t)(y0 + ty + j) * HH + x0 + tx];
    __syncthreads();
    _Float16* dst = Xt + (size_t)img * PLANE_X;
    #pragma unroll
    for (int j = 0; j < 32; j += 8)
        dst[(size_t)(x0 + ty + j) * HH + y0 + tx] = (_Float16)tile[tx][ty + j];
}

// ---------------------------------------------------------------------------
// Pointwise frequency product + CIN reduction:
//   G[b,o][u][v] = sum_c F[b,c][u][v] * W[o][c][u][v]    (complex, fp32 acc)
// Weight (~67 MB) is the dominant stream -> read exactly once, coalesced in v.
// ---------------------------------------------------------------------------
__global__ void pointwise_kernel(const _Float16* __restrict__ F,
                                 const float* __restrict__ W,
                                 _Float16* __restrict__ G) {
    const int v = blockIdx.x * 256 + threadIdx.x;   // 0..511
    const int u = blockIdx.y;                       // 0..511
    const int o = blockIdx.z;                       // 0..7
    const bool inb = (u < NDFT) && (v < NDFT);
    float wr[CIN], wi[CIN];
    if (inb) {
        #pragma unroll
        for (int c = 0; c < CIN; ++c) {
            const float* wp = W + ((((size_t)o * CIN + c) * NDFT + u) * NDFT + v) * 2;
            __builtin_prefetch(wp + 2 * NDFT, 0, 1);   // next-u row of this stream
            wr[c] = wp[0]; wi[c] = wp[1];
        }
    }
    #pragma unroll
    for (int b = 0; b < BB; ++b) {
        float ar = 0.0f, ai = 0.0f;
        if (inb) {
            #pragma unroll
            for (int c = 0; c < CIN; ++c) {
                size_t fb = ((size_t)(b * CIN + c) * 2) * PLANE_F + (size_t)u * NP + v;
                float fr = (float)F[fb], fi = (float)F[fb + PLANE_F];
                ar += fr * wr[c] - fi * wi[c];
                ai += fr * wi[c] + fi * wr[c];
            }
        }
        size_t gb = ((size_t)(b * COUT + o) * 2) * PLANE_G + (size_t)u * NP + v;
        G[gb] = (_Float16)ar; G[gb + PLANE_G] = (_Float16)ai;
    }
}

// ---------------------------------------------------------------------------
// Generic batched two-product GEMM:  C = A1*B1t [+ A2*B2t]
// A: row-major MxK (fp16), B: K-contiguous NxK "Bt" (fp16).
// Per-lane fragment loads use contiguous k order (k = 16*half + s) for both A and B.
// Complex-multiply minus signs are baked into pre-negated twiddle planes,
// so there is no negate/select in the hot loop; DUAL is a template param so
// there is no branch in the hot loop either.
// ---------------------------------------------------------------------------
struct GemmDesc {
    const _Float16* A1; const _Float16* B1;
    const _Float16* A2; const _Float16* B2;
    long long sA, sB;          // batch strides (halves) for A1/A2 and B1/B2
    int lda, ldb, K;
    _Float16* Ch; long long sC; int ldc; int transC;   // fp16 output modes
    float* Cf; long long sCf; int ldcf;                // fp32 output mode
    const float* bias; int biasMod;
    float scale;
};

__device__ __forceinline__ v16h load_a_frag(const _Float16* A, int lda,
                                            int m, int k, int lane) {
    // lane<16: M=lane,   K = k+0..15
    // lane>=16:M=lane-16,K = k+16..31
    const _Float16* p = A + (size_t)(m + (lane & 15)) * lda + k + ((lane >> 4) << 4);
    return *(const v16h*)p;
}

__device__ __forceinline__ v16h load_b_frag(const _Float16* Bt, int ldb,
                                            int n, int k, int lane) {
    // lane<16: N=lane,   K = k+0..15 ; lane>=16: N=lane-16, K = k+16..31
    const _Float16* p = Bt + (size_t)(n + (lane & 15)) * ldb + k + ((lane >> 4) << 4);
    return *(const v16h*)p;
}

#define WMMA_F16(a, b, c) \
    __builtin_amdgcn_wmma_f32_16x16x32_f16(false, (a), false, (b), (short)0, (c), false, false)

template <bool DUAL>
__global__ __launch_bounds__(256) void gemm_kernel(GemmDesc d) {
    const int lane = threadIdx.x & 31;
    const int wave = threadIdx.x >> 5;
    const int mBase = blockIdx.x * 64  + (wave & 1) * 32;   // 2 waves in M
    const int nBase = blockIdx.y * 128 + (wave >> 1) * 32;  // 4 waves in N
    const int z = blockIdx.z;

    const _Float16* A1 = d.A1 + (size_t)z * d.sA;
    const _Float16* B1 = d.B1 + (size_t)z * d.sB;
    const _Float16* A2 = DUAL ? d.A2 + (size_t)z * d.sA : nullptr;
    const _Float16* B2 = DUAL ? d.B2 + (size_t)z * d.sB : nullptr;

    v8f acc[2][2] = {};
    for (int k = 0; k < d.K; k += 32) {
        {
            v16h a0 = load_a_frag(A1, d.lda, mBase,      k, lane);
            v16h a1 = load_a_frag(A1, d.lda, mBase + 16, k, lane);
            v16h b0 = load_b_frag(B1, d.ldb, nBase,      k, lane);
            v16h b1 = load_b_frag(B1, d.ldb, nBase + 16, k, lane);
            acc[0][0] = WMMA_F16(a0, b0, acc[0][0]);
            acc[0][1] = WMMA_F16(a0, b1, acc[0][1]);
            acc[1][0] = WMMA_F16(a1, b0, acc[1][0]);
            acc[1][1] = WMMA_F16(a1, b1, acc[1][1]);
        }
        if (DUAL) {
            v16h a0 = load_a_frag(A2, d.lda, mBase,      k, lane);
            v16h a1 = load_a_frag(A2, d.lda, mBase + 16, k, lane);
            v16h b0 = load_b_frag(B2, d.ldb, nBase,      k, lane);
            v16h b1 = load_b_frag(B2, d.ldb, nBase + 16, k, lane);
            acc[0][0] = WMMA_F16(a0, b0, acc[0][0]);
            acc[0][1] = WMMA_F16(a0, b1, acc[0][1]);
            acc[1][0] = WMMA_F16(a1, b0, acc[1][0]);
            acc[1][1] = WMMA_F16(a1, b1, acc[1][1]);
        }
    }

    // C/D layout: vgpr r, lane L -> (M = r + (L<16?0:8), N = L&15)
    const int mOff = (lane >> 4) << 3;
    const int nl = lane & 15;
    if (d.Cf) {   // fp32 epilogue: scale + per-image bias straight to output
        float* out = d.Cf + (size_t)z * d.sCf;
        const float bv = d.bias ? d.bias[z % d.biasMod] : 0.0f;
        #pragma unroll
        for (int tm = 0; tm < 2; ++tm)
            #pragma unroll
            for (int tn = 0; tn < 2; ++tn)
                #pragma unroll
                for (int r = 0; r < 8; ++r)
                    out[(size_t)(mBase + tm * 16 + mOff + r) * d.ldcf +
                        nBase + tn * 16 + nl] = acc[tm][tn][r] * d.scale + bv;
    } else if (d.transC) {   // fp16 transposed: per-lane contiguous b128 stores
        _Float16* out = d.Ch + (size_t)z * d.sC;
        #pragma unroll
        for (int tm = 0; tm < 2; ++tm)
            #pragma unroll
            for (int tn = 0; tn < 2; ++tn) {
                v8h h;
                #pragma unroll
                for (int r = 0; r < 8; ++r) h[r] = (_Float16)acc[tm][tn][r];
                *(v8h*)(out + (size_t)(nBase + tn * 16 + nl) * d.ldc +
                        mBase + tm * 16 + mOff) = h;
            }
    } else {                 // fp16 row-major
        _Float16* out = d.Ch + (size_t)z * d.sC;
        #pragma unroll
        for (int tm = 0; tm < 2; ++tm)
            #pragma unroll
            for (int tn = 0; tn < 2; ++tn)
                #pragma unroll
                for (int r = 0; r < 8; ++r)
                    out[(size_t)(mBase + tm * 16 + mOff + r) * d.ldc +
                        nBase + tn * 16 + nl] = (_Float16)acc[tm][tn][r];
    }
}

// ---------------------------------------------------------------------------
extern "C" void kernel_launch(void* const* d_in, const int* in_sizes, int n_in,
                              void* d_out, int out_size, void* d_ws, size_t ws_size,
                              hipStream_t stream) {
    const float* im     = (const float*)d_in[0];   // (4,4,256,256)
    const float* weight = (const float*)d_in[1];   // (8,4,511,511,2)
    const float* bias   = (const float*)d_in[2];   // (8,1,1)
    float* out = (float*)d_out;                    // (4,8,256,256)

    // Workspace carve-up (halves). Total = 39,583,744 halves ~= 79.2 MB.
    _Float16* ws  = (_Float16*)d_ws;
    _Float16* Dr  = ws;
    _Float16* Di  = Dr  + PLANE_D;
    _Float16* Dni = Di  + PLANE_D;                      // -Di
    _Float16* Er  = Dni + PLANE_D;
    _Float16* Ei  = Er  + PLANE_E;
    _Float16* Eni = Ei  + PLANE_E;                      // -Ei
    _Float16* Xt  = Eni + PLANE_E;                      // 16 images
    _Float16* T   = Xt  + (size_t)BB * CIN * PLANE_X;   // 16 images x 2 planes
    _Float16* F   = T   + (size_t)BB * CIN * 2 * PLANE_T;
    _Float16* G   = F   + (size_t)BB * CIN * 2 * PLANE_F; // 32 images x 2 planes
    _Float16* Qt  = G   + (size_t)BB * COUT * 2 * PLANE_G;

    twiddle_kernel<<<512, 256, 0, stream>>>(Dr, Di, Dni, Er, Ei, Eni);
    prep_transpose_kernel<<<dim3(8, 8, BB * CIN), dim3(32, 8), 0, stream>>>(im, Xt);

    GemmDesc g;

    // ---- Stage 1: T = D . X  (real X) -> Tr = Dr.Xt, Ti = Di.Xt  [M=512 N=256 K=256]
    g = {}; g.lda = HH; g.ldb = HH; g.K = HH;
    g.sA = 0; g.sB = PLANE_X; g.sC = 2LL * PLANE_T; g.ldc = HH;
    g.A1 = Dr; g.B1 = Xt; g.Ch = T;
    gemm_kernel<false><<<dim3(8, 2, BB * CIN), 256, 0, stream>>>(g);
    g.A1 = Di; g.Ch = T + PLANE_T;
    gemm_kernel<false><<<dim3(8, 2, BB * CIN), 256, 0, stream>>>(g);

    // ---- Stage 2: F = T . D^T  [M=512 N=512 K=256]; Bt = Dfwd itself
    g = {}; g.lda = HH; g.ldb = HH; g.K = HH;
    g.sA = 2LL * PLANE_T; g.sB = 0; g.sC = 2LL * PLANE_F; g.ldc = NP;
    g.A1 = T; g.B1 = Dr; g.A2 = T + PLANE_T; g.B2 = Dni; g.Ch = F;      // Fr = Tr.Dr - Ti.Di
    gemm_kernel<true><<<dim3(8, 4, BB * CIN), 256, 0, stream>>>(g);
    g.B1 = Di; g.B2 = Dr; g.Ch = F + PLANE_F;                           // Fi = Tr.Di + Ti.Dr
    gemm_kernel<true><<<dim3(8, 4, BB * CIN), 256, 0, stream>>>(g);

    // ---- Stage 3: G[b,o] = sum_c F[b,c] * W[o,c]  (~67 MB weight stream)
    pointwise_kernel<<<dim3(2, NP, COUT), 256, 0, stream>>>(F, weight, G);

    // ---- Stage 4: Q = G . E^T, stored transposed Qt[x][u]  [M=512 N=256 K=512]
    g = {}; g.lda = NP; g.ldb = NP; g.K = NP;
    g.sA = 2LL * PLANE_G; g.sB = 0; g.sC = 2LL * PLANE_Q; g.ldc = NP; g.transC = 1;
    g.A1 = G; g.B1 = Er; g.A2 = G + PLANE_G; g.B2 = Eni; g.Ch = Qt;     // Qr = Gr.Er - Gi.Ei
    gemm_kernel<true><<<dim3(8, 2, BB * COUT), 256, 0, stream>>>(g);
    g.B1 = Ei; g.B2 = Er; g.Ch = Qt + PLANE_Q;                          // Qi = Gr.Ei + Gi.Er
    gemm_kernel<true><<<dim3(8, 2, BB * COUT), 256, 0, stream>>>(g);

    // ---- Stage 5: out = Re(E . Q) * s + bias[o]  [M=256 N=256 K=512]
    g = {}; g.lda = NP; g.ldb = NP; g.K = NP;
    g.sA = 0; g.sB = 2LL * PLANE_Q;
    g.A1 = Er; g.B1 = Qt; g.A2 = Eni; g.B2 = Qt + PLANE_Q;              // Er.Qr - Ei.Qi
    g.Cf = out; g.sCf = (long long)HH * HH; g.ldcf = HH;
    g.bias = bias; g.biasMod = COUT;
    g.scale = 1024.0f / (511.0f * 511.0f);   // undo 2x (1/32) in E, apply 1/N^2
    gemm_kernel<true><<<dim3(4, 2, BB * COUT), 256, 0, stream>>>(g);
}
